// MoE_37056977830134
// MI455X (gfx1250) — compile-verified, matrix operand order checked by reading, not run on hardware
//
#include <hip/hip_runtime.h>
#include <math.h>
#include <stdint.h>

// Problem constants (from reference): B=4, S=4096, D=512, E=8, K=2, H=2048
#define T_TOK 16384
#define DDIM  512
#define EEXP  8
#define KTOP  2
#define HDIM  2048

#if defined(__has_builtin)
#  if __has_builtin(__builtin_amdgcn_global_load_async_to_lds_b128)
#    define HAVE_ASYNC_LDS 1
#  endif
#endif
#ifndef HAVE_ASYNC_LDS
#  define HAVE_ASYNC_LDS 0
#endif

typedef __attribute__((ext_vector_type(16))) __bf16 v16bf;
typedef __attribute__((ext_vector_type(8)))  float  v8f;
typedef __attribute__((ext_vector_type(4)))  int    v4i;

typedef __attribute__((address_space(1))) v4i* gptr_v4i;   // global
typedef __attribute__((address_space(3))) v4i* lptr_v4i;   // LDS

union Frag16 { v16bf v; unsigned int u[8]; };

__device__ __forceinline__ unsigned short f2bf(float f) {
  unsigned int u = __float_as_uint(f);
  unsigned int r = (u + 0x7FFFu + ((u >> 16) & 1u)) >> 16;   // round-nearest-even
  return (unsigned short)r;
}

// K-pair start for VGPR v of a 16-bit A/B fragment (ISA 7.12.2, wave32):
// lanes 0-15 (half=0): VGPR0..3 -> K=0,2,4,6 ; VGPR4..7 -> K=16,18,20,22
// lanes 16-31 (half=1): +8
__device__ __forceinline__ int kpair(int v, int half) {
  return (v < 4 ? v * 2 : 16 + (v - 4) * 2) + half * 8;
}

__device__ __forceinline__ float gelu_exact(float v) {
  return 0.5f * v * (1.0f + erff(v * 0.70710678118654752f));
}

// 16B global -> LDS copy: async (ASYNCcnt-tracked, no VGPR round trip) if the
// toolchain exposes the gfx1250 builtin, else a plain vector copy.
__device__ __forceinline__ void copy_b128_to_lds(const void* gsrc, void* lds_dst) {
#if HAVE_ASYNC_LDS
  __builtin_amdgcn_global_load_async_to_lds_b128(
      (gptr_v4i)(uintptr_t)gsrc,
      (lptr_v4i)(uintptr_t)lds_dst,
      0, 0);
#else
  *(uint4*)lds_dst = *(const uint4*)gsrc;
#endif
}

__device__ __forceinline__ void wait_async_all() {
#if HAVE_ASYNC_LDS
#  if __has_builtin(__builtin_amdgcn_s_wait_asynccnt)
  __builtin_amdgcn_s_wait_asynccnt(0);
#  else
  asm volatile("s_wait_asynccnt 0x0" ::: "memory");
#  endif
#endif
}

// ---------------------------------------------------------------- init ------
__global__ void moe_init(float* out, int* ctrl, long long n_out) {
  long long i = (long long)blockIdx.x * blockDim.x + threadIdx.x;
  if (i < n_out) out[i] = 0.0f;
  if (blockIdx.x == 0 && threadIdx.x < 32) ctrl[threadIdx.x] = 0;
}

// ------------------------------------------------------------ fp32->bf16 ---
__global__ void cvt_bf16(const float* __restrict__ src,
                         unsigned short* __restrict__ dst, long long n) {
  long long i = (long long)blockIdx.x * blockDim.x + threadIdx.x;
  if (i < n) dst[i] = f2bf(src[i]);
}

// --------------------------- fp32 [R][C] -> bf16 [C][R] (per expert) -------
// block (32,8); 32x32 tile via padded LDS. grid = (C/32, R/32, E).
__global__ void transpose_cvt_bf16(const float* __restrict__ src,
                                   unsigned short* __restrict__ dst,
                                   int R, int C) {
  __shared__ float tile[32][33];
  const int e = blockIdx.z;
  const float* in = src + (long long)e * R * C;
  unsigned short* outp = dst + (long long)e * R * C;
  const int c0 = blockIdx.x * 32, r0 = blockIdx.y * 32;
  const int tx = threadIdx.x, ty = threadIdx.y;
#pragma unroll
  for (int j = 0; j < 32; j += 8)
    tile[ty + j][tx] = in[(long long)(r0 + ty + j) * C + c0 + tx];
  __syncthreads();
#pragma unroll
  for (int j = 0; j < 32; j += 8)
    outp[(long long)(c0 + ty + j) * R + r0 + tx] = f2bf(tile[tx][ty + j]);
}

// ---------------------------------------------------------------- gating ---
__global__ void gate_kernel(const float* __restrict__ x,
                            const float* __restrict__ wg,
                            int* __restrict__ topi, float* __restrict__ topw,
                            int* __restrict__ counts) {
  int t = blockIdx.x * blockDim.x + threadIdx.x;
  if (t >= T_TOK) return;
  const float* xr = x + (long long)t * DDIM;
  float acc[EEXP];
#pragma unroll
  for (int e = 0; e < EEXP; ++e) acc[e] = 0.0f;
  for (int d = 0; d < DDIM; ++d) {
    float xv = xr[d];
#pragma unroll
    for (int e = 0; e < EEXP; ++e) acc[e] += xv * wg[d * EEXP + e];
  }
  int i0 = 0; float v0 = acc[0];
#pragma unroll
  for (int e = 1; e < EEXP; ++e) if (acc[e] > v0) { v0 = acc[e]; i0 = e; }
  int i1 = -1; float v1 = -3.4e38f;
#pragma unroll
  for (int e = 0; e < EEXP; ++e)
    if (e != i0 && acc[e] > v1) { v1 = acc[e]; i1 = e; }
  float w0 = 1.0f / (1.0f + expf(v1 - v0));   // softmax over {v0, v1}
  float w1 = 1.0f - w0;
  topi[t * 2 + 0] = i0; topi[t * 2 + 1] = i1;
  topw[t * 2 + 0] = w0; topw[t * 2 + 1] = w1;
  atomicAdd(&counts[i0], 1);
  atomicAdd(&counts[i1], 1);
}

// --------------------------------------------- offsets + lb_loss (1 thread) -
__global__ void finalize_kernel(const int* __restrict__ counts,
                                int* __restrict__ offsets,
                                float* __restrict__ lb_out) {
  if (threadIdx.x != 0 || blockIdx.x != 0) return;
  int off = 0;
  float mean = 0.0f;
  for (int e = 0; e < EEXP; ++e) {
    offsets[e] = off;
    off += counts[e];
    mean += (float)counts[e];
  }
  mean *= (1.0f / EEXP);
  float var = 0.0f;
  for (int e = 0; e < EEXP; ++e) {
    float d = (float)counts[e] - mean;
    var += d * d;
  }
  var *= (1.0f / (EEXP - 1));                 // ddof=1
  *lb_out = var / (mean * mean + 1e-6f);
}

// -------------------------------------------------------------- dispatch ---
__global__ void scatter_kernel(const int* __restrict__ topi,
                               const float* __restrict__ topw,
                               const int* __restrict__ offsets,
                               int* __restrict__ cursors,
                               int* __restrict__ token_list,
                               float* __restrict__ gate_list) {
  int t = blockIdx.x * blockDim.x + threadIdx.x;
  if (t >= T_TOK) return;
#pragma unroll
  for (int k = 0; k < KTOP; ++k) {
    int e = topi[t * 2 + k];
    int pos = atomicAdd(&cursors[e], 1);
    int s = offsets[e] + pos;
    token_list[s] = t;
    gate_list[s]  = topw[t * 2 + k];
  }
}

// --------------------------------------- expert GEMM 1: h = gelu(x@W1 + b1) -
// block = 256 threads (8 waves); block tile = 32 tokens x 256 H;
// each wave owns a 32x32 patch -> 4 x v_wmma_f32_16x16x32_bf16 per k-step.
// W1 is pre-transposed to [E][H][D] so B tiles are linear row segments.
__global__ void __launch_bounds__(256)
ffn1_kernel(const unsigned short* __restrict__ xbf,
            const unsigned short* __restrict__ w1t,   // [E][H][D] bf16
            const float* __restrict__ b1,
            const int* __restrict__ counts,
            const int* __restrict__ offsets,
            const int* __restrict__ token_list,
            unsigned short* __restrict__ hidden) {
  const int e  = blockIdx.z;
  const int mt = blockIdx.y;                // 32-token tile
  const int nt = blockIdx.x;                // 256-wide H tile
  const int cnt = counts[e];
  if (mt * 32 >= cnt) return;
  const int eoff = offsets[e];

  __shared__ __align__(16) unsigned short As[32 * 512];  // [m][k], 32KB
  __shared__ __align__(16) unsigned short Bs[256 * 32];  // [n][k], 16KB
  __shared__ int toks[32];

  const int tid  = threadIdx.x;
  const int lane = tid & 31;
  const int wv   = tid >> 5;
  const int half = lane >> 4;
  const int mrow = lane & 15;
  const int nloc = lane & 15;
  const int hbase = nt * 256;

  if (tid < 32) {
    int m = mt * 32 + tid;
    toks[tid] = (m < cnt) ? token_list[eoff + m] : -1;
  }
  __syncthreads();

  // ---- stage A once: 32 gathered rows x 512 K (async b128 per lane) ----
  for (int v = tid; v < 32 * 64; v += 256) {     // 64 x 16B per row
    int m = v >> 6, c = v & 63;
    int t = toks[m];
    if (t >= 0)
      copy_b128_to_lds(xbf + (long long)t * DDIM + c * 8, (uint4*)As + v);
    else
      ((uint4*)As)[v] = make_uint4(0u, 0u, 0u, 0u);
  }
  wait_async_all();
  __syncthreads();

  v8f acc00 = {0.f,0.f,0.f,0.f,0.f,0.f,0.f,0.f};
  v8f acc01 = acc00, acc10 = acc00, acc11 = acc00;

  for (int s = 0; s < DDIM / 32; ++s) {
    const int k0 = s * 32;
    // ---- stage B: 256 rows x 32 K, each row a 64B linear segment of w1T ----
#pragma unroll
    for (int u = 0; u < 4; ++u) {
      int id = tid + u * 256;                  // 0..1023
      int n  = id >> 2;
      int kc = (id & 3) * 8;
      copy_b128_to_lds(w1t + ((long long)e * HDIM + hbase + n) * DDIM + k0 + kc,
                       &Bs[n * 32 + kc]);
    }
    if (s + 1 < DDIM / 32)   // -> global_prefetch_b8 on next K slice
      __builtin_prefetch(&w1t[((long long)e * HDIM + hbase + tid) * DDIM + k0 + 32], 0, 1);
    wait_async_all();
    __syncthreads();

    Frag16 a0, a1, b0, b1;
#pragma unroll
    for (int v = 0; v < 8; ++v) {
      int kp = kpair(v, half);                 // even -> aligned dwords
      a0.u[v] = *(const unsigned int*)&As[(mrow +  0) * 512 + k0 + kp];
      a1.u[v] = *(const unsigned int*)&As[(mrow + 16) * 512 + k0 + kp];
      b0.u[v] = *(const unsigned int*)&Bs[(wv * 32 + nloc +  0) * 32 + kp];
      b1.u[v] = *(const unsigned int*)&Bs[(wv * 32 + nloc + 16) * 32 + kp];
    }
    acc00 = __builtin_amdgcn_wmma_f32_16x16x32_bf16(false, a0.v, false, b0.v, (short)0, acc00, false, false);
    acc01 = __builtin_amdgcn_wmma_f32_16x16x32_bf16(false, a0.v, false, b1.v, (short)0, acc01, false, false);
    acc10 = __builtin_amdgcn_wmma_f32_16x16x32_bf16(false, a1.v, false, b0.v, (short)0, acc10, false, false);
    acc11 = __builtin_amdgcn_wmma_f32_16x16x32_bf16(false, a1.v, false, b1.v, (short)0, acc11, false, false);
    __syncthreads();
  }

  // ---- epilogue: gelu(acc + b1) -> bf16 hidden ----
#pragma unroll
  for (int q = 0; q < 4; ++q) {
    const v8f& a = (q == 0) ? acc00 : (q == 1) ? acc01 : (q == 2) ? acc10 : acc11;
    int msub = (q >> 1) * 16, nsub = (q & 1) * 16;
    int n = hbase + wv * 32 + nloc + nsub;
    float bias = b1[e * HDIM + n];
#pragma unroll
    for (int r = 0; r < 8; ++r) {
      int m  = r + 8 * half + msub;            // C/D layout: VGPR r, lane half
      int gm = mt * 32 + m;
      if (gm < cnt)
        hidden[(long long)(eoff + gm) * HDIM + n] = f2bf(gelu_exact(a[r] + bias));
    }
  }
}

// --------------------------- expert GEMM 2: out += gate * (h@W2 + b2) ------
// block tile = 32 slots x 256 D; K=2048 staged in 512-wide A chunks.
// W2 pre-transposed to [E][D][H] so B tiles are linear row segments.
__global__ void __launch_bounds__(256)
ffn2_kernel(const unsigned short* __restrict__ hidden,
            const unsigned short* __restrict__ w2t,   // [E][D][H] bf16
            const float* __restrict__ b2,
            const int* __restrict__ counts,
            const int* __restrict__ offsets,
            const int* __restrict__ token_list,
            const float* __restrict__ gate_list,
            float* __restrict__ out) {
  const int e  = blockIdx.z;
  const int mt = blockIdx.y;
  const int nt = blockIdx.x;                // 256-wide D tile (2 total)
  const int cnt = counts[e];
  if (mt * 32 >= cnt) return;
  const int eoff = offsets[e];

  __shared__ __align__(16) unsigned short As[32 * 512];
  __shared__ __align__(16) unsigned short Bs[256 * 32];

  const int tid  = threadIdx.x;
  const int lane = tid & 31;
  const int wv   = tid >> 5;
  const int half = lane >> 4;
  const int mrow = lane & 15;
  const int nloc = lane & 15;
  const int dbase = nt * 256;

  v8f acc00 = {0.f,0.f,0.f,0.f,0.f,0.f,0.f,0.f};
  v8f acc01 = acc00, acc10 = acc00, acc11 = acc00;

  for (int ch = 0; ch < HDIM / 512; ++ch) {       // 4 K-chunks of 512
    // ---- stage A chunk: 32 contiguous hidden rows x 512 K (async b128) ----
    for (int v = tid; v < 32 * 64; v += 256) {
      int m = v >> 6, c = v & 63;
      int gm = mt * 32 + m;
      if (gm < cnt)
        copy_b128_to_lds(hidden + (long long)(eoff + gm) * HDIM + ch * 512 + c * 8,
                         (uint4*)As + v);
      else
        ((uint4*)As)[v] = make_uint4(0u, 0u, 0u, 0u);
    }
    wait_async_all();
    __syncthreads();

    for (int s = 0; s < 16; ++s) {
      const int k0g = ch * 512 + s * 32;          // global K
      const int k0l = s * 32;                     // K within A chunk
#pragma unroll
      for (int u = 0; u < 4; ++u) {
        int id = tid + u * 256;
        int n  = id >> 2;
        int kc = (id & 3) * 8;
        copy_b128_to_lds(w2t + ((long long)e * DDIM + dbase + n) * HDIM + k0g + kc,
                         &Bs[n * 32 + kc]);
      }
      if (k0g + 32 < HDIM)
        __builtin_prefetch(&w2t[((long long)e * DDIM + dbase + tid) * HDIM + k0g + 32], 0, 1);
      wait_async_all();
      __syncthreads();

      Frag16 a0, a1, b0, b1;
#pragma unroll
      for (int v = 0; v < 8; ++v) {
        int kp = kpair(v, half);
        a0.u[v] = *(const unsigned int*)&As[(mrow +  0) * 512 + k0l + kp];
        a1.u[v] = *(const unsigned int*)&As[(mrow + 16) * 512 + k0l + kp];
        b0.u[v] = *(const unsigned int*)&Bs[(wv * 32 + nloc +  0) * 32 + kp];
        b1.u[v] = *(const unsigned int*)&Bs[(wv * 32 + nloc + 16) * 32 + kp];
      }
      acc00 = __builtin_amdgcn_wmma_f32_16x16x32_bf16(false, a0.v, false, b0.v, (short)0, acc00, false, false);
      acc01 = __builtin_amdgcn_wmma_f32_16x16x32_bf16(false, a0.v, false, b1.v, (short)0, acc01, false, false);
      acc10 = __builtin_amdgcn_wmma_f32_16x16x32_bf16(false, a1.v, false, b0.v, (short)0, acc10, false, false);
      acc11 = __builtin_amdgcn_wmma_f32_16x16x32_bf16(false, a1.v, false, b1.v, (short)0, acc11, false, false);
      __syncthreads();
    }
  }

  // ---- epilogue: out[token] += gate * (acc + b2) ----
#pragma unroll
  for (int q = 0; q < 4; ++q) {
    const v8f& a = (q == 0) ? acc00 : (q == 1) ? acc01 : (q == 2) ? acc10 : acc11;
    int msub = (q >> 1) * 16, nsub = (q & 1) * 16;
    int d = dbase + wv * 32 + nloc + nsub;
    float bias = b2[e * DDIM + d];
#pragma unroll
    for (int r = 0; r < 8; ++r) {
      int m  = r + 8 * half + msub;
      int gm = mt * 32 + m;
      if (gm < cnt) {
        int   t = token_list[eoff + gm];
        float g = gate_list[eoff + gm];
        atomicAdd(&out[(long long)t * DDIM + d], g * (a[r] + bias));
      }
    }
  }
}

// ---------------------------------------------------------------- launch ---
extern "C" void kernel_launch(void* const* d_in, const int* in_sizes, int n_in,
                              void* d_out, int out_size, void* d_ws, size_t ws_size,
                              hipStream_t stream) {
  const float* x  = (const float*)d_in[0];   // [T, D]
  const float* wg = (const float*)d_in[1];   // [D, E]
  const float* W1 = (const float*)d_in[2];   // [E, D, H]
  const float* b1 = (const float*)d_in[3];   // [E, H]
  const float* W2 = (const float*)d_in[4];   // [E, H, D]
  const float* b2 = (const float*)d_in[5];   // [E, D]
  float* out = (float*)d_out;                // T*D floats + 1 (lb_loss)

  const long long nX  = (long long)T_TOK * DDIM;          // 8.4M
  const long long nW1 = (long long)EEXP * DDIM * HDIM;    // 8.4M
  const long long nW2 = (long long)EEXP * HDIM * DDIM;    // 8.4M
  const long long nH  = (long long)T_TOK * KTOP * HDIM;   // 67M

  // ---- workspace carve (4-byte sections first; bf16 blocks 16B-aligned) ----
  char* p = (char*)d_ws;
  int*   ctrl       = (int*)p;            p += 32 * sizeof(int);   // counts[8] cursors[8] offsets[8]
  int*   topi       = (int*)p;            p += (long long)T_TOK * 2 * sizeof(int);
  float* topw       = (float*)p;          p += (long long)T_TOK * 2 * sizeof(float);
  int*   token_list = (int*)p;            p += (long long)T_TOK * KTOP * sizeof(int);
  float* gate_list  = (float*)p;          p += (long long)T_TOK * KTOP * sizeof(float);
  unsigned short* xbf    = (unsigned short*)p; p += nX  * 2;
  unsigned short* w1t    = (unsigned short*)p; p += nW1 * 2;   // [E][H][D]
  unsigned short* w2t    = (unsigned short*)p; p += nW2 * 2;   // [E][D][H]
  unsigned short* hidden = (unsigned short*)p; p += nH  * 2;

  int* counts  = ctrl;
  int* cursors = ctrl + 8;
  int* offsets = ctrl + 16;

  // 1) zero output + control
  moe_init<<<(unsigned)((nX + 255) / 256), 256, 0, stream>>>(out, ctrl, nX);
  // 2) x: fp32 -> bf16 ; weights: fp32 -> bf16 with transpose (N-major rows)
  cvt_bf16<<<(unsigned)((nX + 255) / 256), 256, 0, stream>>>(x, xbf, nX);
  {
    dim3 tb(32, 8);
    transpose_cvt_bf16<<<dim3(HDIM / 32, DDIM / 32, EEXP), tb, 0, stream>>>(
        W1, w1t, DDIM, HDIM);   // [E][D][H] -> [E][H][D]
    transpose_cvt_bf16<<<dim3(DDIM / 32, HDIM / 32, EEXP), tb, 0, stream>>>(
        W2, w2t, HDIM, DDIM);   // [E][H][D] -> [E][D][H]
  }
  // 3) gating (logits, top-2, softmax, expert loads)
  gate_kernel<<<T_TOK / 256, 256, 0, stream>>>(x, wg, topi, topw, counts);
  // 4) offsets + lb_loss
  finalize_kernel<<<1, 32, 0, stream>>>(counts, offsets, out + nX);
  // 5) build per-expert token lists
  scatter_kernel<<<T_TOK / 256, 256, 0, stream>>>(topi, topw, offsets, cursors,
                                                  token_list, gate_list);
  // 6) expert GEMM 1 + GELU  (8 H-tiles x up-to-512 M-tiles x 8 experts)
  {
    dim3 grid(HDIM / 256, T_TOK / 32, EEXP);
    ffn1_kernel<<<grid, 256, 0, stream>>>(xbf, w1t, b1, counts, offsets,
                                          token_list, hidden);
  }
  // 7) expert GEMM 2 + gate-weighted scatter-add (2 D-tiles)
  {
    dim3 grid(DDIM / 256, T_TOK / 32, EEXP);
    ffn2_kernel<<<grid, 256, 0, stream>>>(hidden, w2t, b2, counts, offsets,
                                          token_list, gate_list, out);
  }
}
